// KoopmanMlpExtractor_10934986736293
// MI455X (gfx1250) — matrix operand order repaired and checked by reading, not compiled, not run on hardware
//
#include <hip/hip_runtime.h>
#include <hip/hip_bf16.h>

typedef __attribute__((ext_vector_type(16))) __bf16 v16bf;
typedef __attribute__((ext_vector_type(8)))  float  v8f;

#define DDIM       64
#define NEXP       64
#define WAVES      16
#define NTHREADS   512
#define ROWS       256          // batch rows per block
#define XSTRIDE    68           // dword row stride for sigma/x LDS (conflict-free, 16B aligned)
#define SLABSTRIDE 36           // dword row stride for B-operand slabs (conflict-free b128)
#define NCHUNK     128          // K=4096 / 32

union BOp {
  uint4    q[2];
  unsigned u[8];
  v16bf    v;
};

// dst[15:0] = bf16(lo), dst[31:16] = bf16(hi)  -- one VALU op
static __device__ __forceinline__ unsigned bf16pack(float lo, float hi) {
  unsigned d;
  asm("v_cvt_pk_bf16_f32 %0, %1, %2" : "=v"(d) : "v"(lo), "v"(hi));
  return d;
}
// packed bf16 multiply -- one VALU op
static __device__ __forceinline__ unsigned pk_mul_bf16(unsigned a, unsigned b) {
  unsigned d;
  asm("v_pk_mul_bf16 %0, %1, %2" : "=v"(d) : "v"(a), "v"(b));
  return d;
}
static __device__ __forceinline__ unsigned short bfbits(float a) {
  return __builtin_bit_cast(unsigned short, (__bf16)a);
}
static __device__ __forceinline__ v8f wmma_bf16(v16bf a, v16bf b, v8f c) {
  // D = A(16x32 bf16) x B(32x16 bf16) + C(16x16 f32)
  return __builtin_amdgcn_wmma_f32_16x16x32_bf16(false, a, false, b, (short)0, c, false, false);
}

// Stage global W/A chunk -> registers (coalesced 128B-per-row float2 loads).
static __device__ __forceinline__ void stage_gload(const float* __restrict__ gW,
                                                   const float* __restrict__ gA,
                                                   int kc, int t, float2* pre) {
  const int eq = kc >> 1;   // expert index (actor) / i index (critic)
  const int h  = kc & 1;    // 32-wide half of the inner 64
  #pragma unroll
  for (int s = 0; s < 2; ++s) {
    const int q = t + s * NTHREADS;   // 0..1023 bf16-pair slots
    const int n = q >> 4;             // 0..63  (N index: o for W, e for A)
    const int m = q & 15;             // 0..15  (k pair)
    // W slab element (k=2m, n) = W[eq, n, 32h+2m]
    pre[s]     = *(const float2*)(gW + ((eq << 12) + (n << 6) + (h << 5) + (m << 1)));
    // A slab element (k=2m, n) = A[n, eq, 32h+2m]
    pre[2 + s] = *(const float2*)(gA + ((n << 12) + (eq << 6) + (h << 5) + (m << 1)));
  }
}
static __device__ __forceinline__ void stage_sstore(unsigned* bufW, unsigned* bufA,
                                                    int t, const float2* pre) {
  #pragma unroll
  for (int s = 0; s < 2; ++s) {
    const int q = t + s * NTHREADS;
    const int n = q >> 4;
    const int m = q & 15;
    bufW[n * SLABSTRIDE + m] = bf16pack(pre[s].x,     pre[s].y);
    bufA[n * SLABSTRIDE + m] = bf16pack(pre[2 + s].x, pre[2 + s].y);
  }
}

__global__ __launch_bounds__(NTHREADS, 1)
void koopman_fused(const float* __restrict__ gx,
                   const float* __restrict__ gW,   // [E,O,D]
                   const float* __restrict__ gA,   // [E,D,D]
                   const float* __restrict__ gb,   // [E,D]
                   const float* __restrict__ gc,   // [E]
                   const float* __restrict__ gWc,  // [E,D]
                   const float* __restrict__ gbc,  // [E]
                   float* __restrict__ actor,      // [B,O]
                   float* __restrict__ critic)     // [B]
{
  __shared__ __align__(16) unsigned s_sx[ROWS * XSTRIDE];          // lo16=bf16(sigma), hi16=bf16(x)
  __shared__ __align__(16) unsigned s_slab[2][2][64 * SLABSTRIDE]; // [buf][W/A] B-operand slabs

  const int t  = threadIdx.x;
  const int w  = t >> 5;
  const int l  = t & 31;
  const int lr = l & 15;
  const int hl = l >> 4;
  const int rowBase = blockIdx.x * ROWS;
  const int wrow = w << 4;

  // ---- stage x (bf16) into hi bits of s_sx, coalesced float4 loads ----
  #pragma unroll
  for (int it = 0; it < (ROWS * DDIM / 4) / NTHREADS; ++it) {
    const int i  = t + it * NTHREADS;          // float4 index 0..4095
    const int r  = i >> 4;
    const int c4 = (i & 15) << 2;
    const float4 vx = *(const float4*)(gx + (size_t)(rowBase + r) * DDIM + c4);
    s_sx[r * XSTRIDE + c4 + 0] = bf16pack(0.0f, vx.x);
    s_sx[r * XSTRIDE + c4 + 1] = bf16pack(0.0f, vx.y);
    s_sx[r * XSTRIDE + c4 + 2] = bf16pack(0.0f, vx.z);
    s_sx[r * XSTRIDE + c4 + 3] = bf16pack(0.0f, vx.w);
  }
  __syncthreads();

  // ---- build X in WMMA A-layout (bf16), both 32-wide K halves ----
  BOp xa0, xa1;
  {
    const float* xr = gx + (size_t)(rowBase + wrow + lr) * DDIM;  // row M = lane&15
    #pragma unroll
    for (int v = 0; v < 8; ++v) {
      const int k = (v << 1) + (hl << 4); // contiguous K layout: k = 16*hl + 2v
      xa0.u[v] = bf16pack(xr[k],      xr[k + 1]);
      xa1.u[v] = bf16pack(xr[32 + k], xr[32 + k + 1]);
    }
  }

  // ---- sigma = sigmoid(x@Wc^T + bc); qd init = x@b^T + c ----
  v8f sC[4], qdC[4], acC[4];
  const v8f vzero = {0.f, 0.f, 0.f, 0.f, 0.f, 0.f, 0.f, 0.f};
  #pragma unroll
  for (int nt = 0; nt < 4; ++nt) { sC[nt] = vzero; qdC[nt] = vzero; acC[nt] = vzero; }

  #pragma unroll
  for (int nt = 0; nt < 4; ++nt) {
    const int e = lr + (nt << 4);              // N index = expert
    #pragma unroll
    for (int h = 0; h < 2; ++h) {
      BOp bw, bb;
      #pragma unroll
      for (int v = 0; v < 8; ++v) {
        const int d = (hl << 4) + (v << 1) + (h << 5);   // contiguous K layout, same as xa0/xa1
        const float2 pw = *(const float2*)(gWc + e * DDIM + d);
        const float2 pb = *(const float2*)(gb  + e * DDIM + d);
        bw.u[v] = bf16pack(pw.x, pw.y);
        bb.u[v] = bf16pack(pb.x, pb.y);
      }
      sC[nt]  = wmma_bf16(h ? xa1.v : xa0.v, bw.v, sC[nt]);
      qdC[nt] = wmma_bf16(h ? xa1.v : xa0.v, bb.v, qdC[nt]);
    }
    const float bcv = gbc[e];
    const float cv  = gc[e];
    #pragma unroll
    for (int v = 0; v < 8; ++v) {
      const float s = 1.0f / (1.0f + __expf(-(sC[nt][v] + bcv)));
      sC[nt][v] = s;
      qdC[nt][v] += cv;
      const int row = wrow + v + (hl << 3);
      const int idx = row * XSTRIDE + e;
      s_sx[idx] = (s_sx[idx] & 0xffff0000u) | (unsigned)bfbits(s);  // unique owner per (row,e)
    }
  }

  // ---- main K=4096 loop: actor += (sigma.x)@Wflat^T ; quad += (x.x)@Aflat^T ----
  float2 pre[4];
  stage_gload(gW, gA, 0, t, pre);
  __syncthreads();                              // sigma writes + x staging visible
  stage_sstore(&s_slab[0][0][0], &s_slab[0][1][0], t, pre);
  __syncthreads();

  const int srow = (wrow + lr) * XSTRIDE;

  #pragma unroll 2
  for (int kc = 0; kc < NCHUNK; ++kc) {
    const int cb = kc & 1;
    if (kc + 1 < NCHUNK) stage_gload(gW, gA, kc + 1, t, pre);   // prefetch next chunk

    // per-lane row scalars for this chunk: sigma[row, e=kc>>1] and x[row, i=kc>>1]
    const unsigned sx = s_sx[srow + (kc >> 1)];
    const unsigned sdup = __builtin_amdgcn_perm(sx, sx, 0x01000100u); // dup lo16 (sigma)
    const unsigned xdup = __builtin_amdgcn_perm(sx, sx, 0x03020302u); // dup hi16 (x_i)
    const BOp& xah = (kc & 1) ? xa1 : xa0;

    BOp xs, xx;                                  // modulated A operands (v_pk_mul_bf16)
    #pragma unroll
    for (int v = 0; v < 8; ++v) {
      xs.u[v] = pk_mul_bf16(xah.u[v], sdup);
      xx.u[v] = pk_mul_bf16(xah.u[v], xdup);
    }

    const unsigned* slW = &s_slab[cb][0][0];
    const unsigned* slA = &s_slab[cb][1][0];
    #pragma unroll
    for (int nt = 0; nt < 4; ++nt) {
      const int off = (lr + (nt << 4)) * SLABSTRIDE + (hl << 3);
      BOp bw, ba;
      bw.q[0] = *(const uint4*)(slW + off);
      bw.q[1] = *(const uint4*)(slW + off + 4);
      ba.q[0] = *(const uint4*)(slA + off);
      ba.q[1] = *(const uint4*)(slA + off + 4);
      acC[nt] = wmma_bf16(xs.v, bw.v, acC[nt]);
      qdC[nt] = wmma_bf16(xx.v, ba.v, qdC[nt]);
    }

    if (kc + 1 < NCHUNK)
      stage_sstore(&s_slab[cb ^ 1][0][0], &s_slab[cb ^ 1][1][0], t, pre);
    __syncthreads();
  }

  // ---- critic: row-sum of sigma .* (quad + x@b^T + c), 16-lane butterfly ----
  #pragma unroll
  for (int v = 0; v < 8; ++v) {
    float acc = 0.0f;
    #pragma unroll
    for (int nt = 0; nt < 4; ++nt) acc += sC[nt][v] * qdC[nt][v];
    acc += __shfl_xor(acc, 1, 32);
    acc += __shfl_xor(acc, 2, 32);
    acc += __shfl_xor(acc, 4, 32);
    acc += __shfl_xor(acc, 8, 32);
    if (lr == 0) critic[rowBase + wrow + (hl << 3) + v] = acc;
  }

  // ---- actor store (C-layout -> [B,64]) ----
  #pragma unroll
  for (int nt = 0; nt < 4; ++nt) {
    #pragma unroll
    for (int v = 0; v < 8; ++v) {
      const int row = rowBase + wrow + v + (hl << 3);
      actor[(size_t)row * 64 + lr + (nt << 4)] = acC[nt][v];
    }
  }
}

extern "C" void kernel_launch(void* const* d_in, const int* in_sizes, int n_in,
                              void* d_out, int out_size, void* d_ws, size_t ws_size,
                              hipStream_t stream) {
  (void)n_in; (void)out_size; (void)d_ws; (void)ws_size;
  const float* x  = (const float*)d_in[0];
  const float* W  = (const float*)d_in[1];
  const float* A  = (const float*)d_in[2];
  const float* b  = (const float*)d_in[3];
  const float* c  = (const float*)d_in[4];
  const float* Wc = (const float*)d_in[5];
  const float* bc = (const float*)d_in[6];
  const int B = in_sizes[0] / DDIM;            // 65536
  float* actor  = (float*)d_out;
  float* critic = (float*)d_out + (size_t)B * DDIM;
  dim3 grid(B / ROWS), block(NTHREADS);
  koopman_fused<<<grid, block, 0, stream>>>(x, W, A, b, c, Wc, bc, actor, critic);
}
